// AttentionModule_75849122447705
// MI455X (gfx1250) — matrix operand order repaired and checked by reading, not compiled, not run on hardware
//
#include <hip/hip_runtime.h>
#include <hip/hip_bf16.h>

#define NEG_INF (-1e30f)

typedef __attribute__((ext_vector_type(16))) __bf16   bf16x16;
typedef __attribute__((ext_vector_type(8)))  float    f32x8;
typedef __attribute__((ext_vector_type(8)))  unsigned uint8v;

// fp32 x2 -> packed bf16 dword. Round-half-up (+0x8000) then splice the two
// high halves with a single v_perm_b32: 3 VALU ops per pair total.
// (Differs from RNE only on exact ties; the cvt_pk_bf16 builtin is absent on
// this toolchain, and the manual RNE sequence cost ~7 VALU ops per pair.)
static __device__ __forceinline__ unsigned pk_bf16(float lo, float hi) {
  unsigned a = __builtin_bit_cast(unsigned, lo) + 0x8000u;
  unsigned b = __builtin_bit_cast(unsigned, hi) + 0x8000u;
  // dst = { b[31:24], b[23:16], a[31:24], a[23:16] }  (sel bytes 7,6,3,2)
  return __builtin_amdgcn_perm(b, a, 0x07060302u);
}

static __device__ __forceinline__ unsigned short bf16of(float f) {
  unsigned a = __builtin_bit_cast(unsigned, f) + 0x8000u;
  return (unsigned short)(a >> 16);
}

static __device__ __forceinline__ f32x8 wmma_bf16(uint8v a, uint8v b, f32x8 c) {
  return __builtin_amdgcn_wmma_f32_16x16x32_bf16(
      false, __builtin_bit_cast(bf16x16, a),
      false, __builtin_bit_cast(bf16x16, b),
      (short)0, c, false, false);
}

// Per-wave LDS ordering: LDS ops from one wave complete in order (ISA 7.3);
// the explicit s_wait_dscnt + memory clobber stops compiler reordering and
// drains outstanding DS ops before the dependent fragment loads.
static __device__ __forceinline__ void wave_lds_sync() {
  __asm__ volatile("s_wait_dscnt 0x0" ::: "memory");
}

// ---------------------------------------------------------------------------
// Kernel 1: words_p[b][c][m] = sum_e W[c][e] * words[b][m][e] + bias[c]
// ---------------------------------------------------------------------------
__global__ __launch_bounds__(256)
void wordproj_kernel(const float* __restrict__ words,  // [B][32][768]
                     const float* __restrict__ Wm,     // [256][768]
                     const float* __restrict__ bias,   // [256]
                     float* __restrict__ wp)           // [B][256][32]
{
  const int b    = blockIdx.x;
  const int tid  = threadIdx.x;
  const int lane = tid & 31;
  const int wid  = tid >> 5;
  const int l16  = lane & 15;
  const int hi   = lane >> 4;

  const float* wordsB = words + (size_t)b * 32 * 768;

  f32x8 acc[2][2] = {{{}, {}}, {{}, {}}};

  for (int k0 = 0; k0 < 768; k0 += 32) {
    uint8v afrag[2];
#pragma unroll
    for (int rt = 0; rt < 2; ++rt) {
      const int row = wid * 32 + rt * 16 + l16;
      const float* ap = Wm + (size_t)row * 768 + k0 + hi * 8;
      float4 a0 = ((const float4*)ap)[0];
      float4 a1 = ((const float4*)ap)[1];
      float4 a2 = ((const float4*)(ap + 16))[0];
      float4 a3 = ((const float4*)(ap + 16))[1];
      uint8v f;
      f[0] = pk_bf16(a0.x, a0.y); f[1] = pk_bf16(a0.z, a0.w);
      f[2] = pk_bf16(a1.x, a1.y); f[3] = pk_bf16(a1.z, a1.w);
      f[4] = pk_bf16(a2.x, a2.y); f[5] = pk_bf16(a2.z, a2.w);
      f[6] = pk_bf16(a3.x, a3.y); f[7] = pk_bf16(a3.z, a3.w);
      afrag[rt] = f;
    }
    uint8v bfrag[2];
#pragma unroll
    for (int nt = 0; nt < 2; ++nt) {
      const int m = nt * 16 + l16;
      const float* bp = wordsB + (size_t)m * 768 + k0 + hi * 8;
      float4 b0 = ((const float4*)bp)[0];
      float4 b1 = ((const float4*)bp)[1];
      float4 b2 = ((const float4*)(bp + 16))[0];
      float4 b3 = ((const float4*)(bp + 16))[1];
      uint8v f;
      f[0] = pk_bf16(b0.x, b0.y); f[1] = pk_bf16(b0.z, b0.w);
      f[2] = pk_bf16(b1.x, b1.y); f[3] = pk_bf16(b1.z, b1.w);
      f[4] = pk_bf16(b2.x, b2.y); f[5] = pk_bf16(b2.z, b2.w);
      f[6] = pk_bf16(b3.x, b3.y); f[7] = pk_bf16(b3.z, b3.w);
      bfrag[nt] = f;
    }
#pragma unroll
    for (int rt = 0; rt < 2; ++rt)
#pragma unroll
      for (int nt = 0; nt < 2; ++nt)
        acc[rt][nt] = wmma_bf16(afrag[rt], bfrag[nt], acc[rt][nt]);
  }

  float* wpB = wp + (size_t)b * 256 * 32;
#pragma unroll
  for (int rt = 0; rt < 2; ++rt)
#pragma unroll
    for (int nt = 0; nt < 2; ++nt) {
      const int m = nt * 16 + l16;
#pragma unroll
      for (int v = 0; v < 8; ++v) {
        const int c = wid * 32 + rt * 16 + v + hi * 8;
        wpB[c * 32 + m] = acc[rt][nt][v] + bias[c];
      }
    }
}

// ---------------------------------------------------------------------------
// Kernel 2: fused  S = X @ Wp * scale -> mask -> softmax -> O = P @ Wpᵀ
// grid (32 pixel-blocks, 32 batches); 8 waves/WG, 16 pixels/wave.
// ---------------------------------------------------------------------------
__global__ __launch_bounds__(256)
void attn_kernel(const float* __restrict__ images, // [B][256][4096]
                 const float* __restrict__ wp,     // [B][256][32] fp32
                 const unsigned char* __restrict__ mask, // [B][32]
                 float* __restrict__ outW,         // [B][256][4096]
                 float* __restrict__ outA)         // [B][32][4096]
{
  const int b    = blockIdx.y;
  const int tid  = threadIdx.x;
  const int lane = tid & 31;
  const int wid  = tid >> 5;
  const int l16  = lane & 15;
  const int hi   = lane >> 4;
  const int p0   = blockIdx.x * 128 + wid * 16;

  // Row pitches chosen so fragment b128 reads hit 16 distinct banks.
  constexpr int WMC_P = 264; // shorts/row, [m][c]   (132 dwords, bank step 4)
  constexpr int WCM_P = 40;  // shorts/row, [c][m]   (20 dwords,  bank step 20)
  constexpr int XS_P  = 20;  // dwords/row, per-wave [pix][chan-pair]
  constexpr int PS_P  = 40;  // shorts/row, per-wave [pix][m]

  __shared__ __align__(16) unsigned short wp_mc[32 * WMC_P];
  __shared__ __align__(16) unsigned short wp_cm[256 * WCM_P];
  __shared__ __align__(16) unsigned       xstage[8][16 * XS_P];
  __shared__ __align__(16) unsigned short pstage[8][16 * PS_P];
  __shared__ unsigned maskf[32];

  // ---- stage words_p[b] into LDS in both bf16 layouts (one channel/thread)
  {
    const float* wpB = wp + (size_t)b * 256 * 32;
    const int c = tid;
    const float4* src = (const float4*)(wpB + c * 32);
    unsigned rowu[16];
#pragma unroll
    for (int i = 0; i < 8; ++i) {
      float4 f = src[i];
      rowu[2 * i]     = pk_bf16(f.x, f.y);
      rowu[2 * i + 1] = pk_bf16(f.z, f.w);
    }
    uint4* dst = (uint4*)&wp_cm[c * WCM_P];
#pragma unroll
    for (int j = 0; j < 4; ++j)
      dst[j] = make_uint4(rowu[4 * j], rowu[4 * j + 1], rowu[4 * j + 2], rowu[4 * j + 3]);
#pragma unroll
    for (int m = 0; m < 32; ++m)
      wp_mc[m * WMC_P + c] = (unsigned short)(rowu[m >> 1] >> ((m & 1) * 16));
    if (tid < 32) maskf[tid] = mask[b * 32 + tid];
  }
  __syncthreads();

  // ---- phase 1: S = X @ Wp  (K = 256, 8 steps of 32 channels)
  f32x8 acc0 = {}, acc1 = {};
  const float* imgB = images + (size_t)b * 256 * 4096 + p0;
  unsigned* xs = &xstage[wid][0];

  for (int kc = 0; kc < 256; kc += 32) {
    // lane: channels kc+2*l16, kc+2*l16+1; pixels (hi*8 .. hi*8+7)
    const float* s0p = imgB + (size_t)(kc + 2 * l16) * 4096 + hi * 8;
    const float* s1p = s0p + 4096;
    if (kc + 32 < 256) {
      __builtin_prefetch(s0p + (size_t)32 * 4096, 0, 1);
      __builtin_prefetch(s1p + (size_t)32 * 4096, 0, 1);
    }
    float4 a0 = ((const float4*)s0p)[0], a1 = ((const float4*)s0p)[1];
    float4 b0 = ((const float4*)s1p)[0], b1 = ((const float4*)s1p)[1];
    unsigned d0 = pk_bf16(a0.x, b0.x), d1 = pk_bf16(a0.y, b0.y);
    unsigned d2 = pk_bf16(a0.z, b0.z), d3 = pk_bf16(a0.w, b0.w);
    unsigned d4 = pk_bf16(a1.x, b1.x), d5 = pk_bf16(a1.y, b1.y);
    unsigned d6 = pk_bf16(a1.z, b1.z), d7 = pk_bf16(a1.w, b1.w);
    const int rb = hi * 8;
    xs[(rb + 0) * XS_P + l16] = d0;
    xs[(rb + 1) * XS_P + l16] = d1;
    xs[(rb + 2) * XS_P + l16] = d2;
    xs[(rb + 3) * XS_P + l16] = d3;
    xs[(rb + 4) * XS_P + l16] = d4;
    xs[(rb + 5) * XS_P + l16] = d5;
    xs[(rb + 6) * XS_P + l16] = d6;
    xs[(rb + 7) * XS_P + l16] = d7;
    wave_lds_sync();   // per-wave stage: DS in-order within wave suffices

    // A fragment: lane = pixel row l16; chan pairs {hi*8..hi*8+7}
    const unsigned* xr = &xstage[wid][l16 * XS_P];
    uint4 alo = *(const uint4*)(xr + hi * 8);
    uint4 ahi = *(const uint4*)(xr + hi * 8 + 4);
    uint8v afrag = {alo.x, alo.y, alo.z, alo.w, ahi.x, ahi.y, ahi.z, ahi.w};

    // B fragments from wp_mc: lane = column m, 16 contiguous K (channels)
    const unsigned short* b0p = &wp_mc[l16 * WMC_P + kc + hi * 16];
    const unsigned short* b1p = &wp_mc[(16 + l16) * WMC_P + kc + hi * 16];
    uint4 q00 = ((const uint4*)b0p)[0], q01 = ((const uint4*)b0p)[1];
    uint4 q10 = ((const uint4*)b1p)[0], q11 = ((const uint4*)b1p)[1];
    uint8v bf0 = {q00.x, q00.y, q00.z, q00.w, q01.x, q01.y, q01.z, q01.w};
    uint8v bf1 = {q10.x, q10.y, q10.z, q10.w, q11.x, q11.y, q11.z, q11.w};

    acc0 = wmma_bf16(afrag, bf0, acc0);   // words 0..15
    acc1 = wmma_bf16(afrag, bf1, acc1);   // words 16..31
    __asm__ volatile("" ::: "memory");    // keep next-iter stores behind reads
  }

  // ---- scale + mask + softmax (rows live across 16-lane halves)
  const float scale = 0.0625f;            // 1/sqrt(256)
  const bool msk0 = maskf[l16] != 0;
  const bool msk1 = maskf[16 + l16] != 0;
  f32x8 pr0, pr1;
#pragma unroll
  for (int v = 0; v < 8; ++v) {
    float s0 = msk0 ? NEG_INF : acc0[v] * scale;
    float s1 = msk1 ? NEG_INF : acc1[v] * scale;
    float r = fmaxf(s0, s1);
    r = fmaxf(r, __shfl_xor(r, 1));
    r = fmaxf(r, __shfl_xor(r, 2));
    r = fmaxf(r, __shfl_xor(r, 4));
    r = fmaxf(r, __shfl_xor(r, 8));       // masks < 16: stays in half-wave
    float e0 = __expf(s0 - r);
    float e1 = __expf(s1 - r);
    float s = e0 + e1;
    s += __shfl_xor(s, 1);
    s += __shfl_xor(s, 2);
    s += __shfl_xor(s, 4);
    s += __shfl_xor(s, 8);
    float inv = 1.0f / s;
    pr0[v] = e0 * inv;
    pr1[v] = e1 * inv;
  }

  // ---- store attn_out[b][m][p]: per lane m fixed, 8 consecutive pixels
  {
    float* aout = outA + (size_t)b * 32 * 4096 + p0 + hi * 8;
    float* a0 = aout + (size_t)l16 * 4096;
    float* a1 = aout + (size_t)(16 + l16) * 4096;
    ((float4*)a0)[0] = make_float4(pr0[0], pr0[1], pr0[2], pr0[3]);
    ((float4*)a0)[1] = make_float4(pr0[4], pr0[5], pr0[6], pr0[7]);
    ((float4*)a1)[0] = make_float4(pr1[0], pr1[1], pr1[2], pr1[3]);
    ((float4*)a1)[1] = make_float4(pr1[4], pr1[5], pr1[6], pr1[7]);
  }

  // ---- C-layout -> A-layout transpose of P through per-wave LDS stage
  {
    unsigned short* ps = &pstage[wid][0];
#pragma unroll
    for (int v = 0; v < 8; ++v) {
      const int pix = v + hi * 8;
      ps[pix * PS_P + l16]      = bf16of(pr0[v]);
      ps[pix * PS_P + 16 + l16] = bf16of(pr1[v]);
    }
  }
  wave_lds_sync();

  // ---- phase 2: O[p][c] = P[p][m] @ Wpᵀ[m][c]  (K = 32, one WMMA per ct)
  const unsigned short* prw = &pstage[wid][l16 * PS_P];
  uint4 plo = *(const uint4*)(prw + hi * 16);
  uint4 phi = *(const uint4*)(prw + hi * 16 + 8);
  uint8v pfrag = {plo.x, plo.y, plo.z, plo.w, phi.x, phi.y, phi.z, phi.w};

  float* wout = outW + (size_t)b * 256 * 4096 + p0 + hi * 8;
#pragma unroll
  for (int ct = 0; ct < 16; ++ct) {
    const int c = ct * 16 + l16;
    const unsigned short* bp = &wp_cm[c * WCM_P + hi * 16]; // column c, K = m
    uint4 blo = ((const uint4*)bp)[0];
    uint4 bhi = ((const uint4*)bp)[1];
    uint8v bfrag = {blo.x, blo.y, blo.z, blo.w, bhi.x, bhi.y, bhi.z, bhi.w};
    f32x8 z = {};
    f32x8 o = wmma_bf16(pfrag, bfrag, z);
    float* wc = wout + (size_t)c * 4096;
    ((float4*)wc)[0] = make_float4(o[0], o[1], o[2], o[3]);
    ((float4*)wc)[1] = make_float4(o[4], o[5], o[6], o[7]);
  }
}

// ---------------------------------------------------------------------------
extern "C" void kernel_launch(void* const* d_in, const int* in_sizes, int n_in,
                              void* d_out, int out_size, void* d_ws, size_t ws_size,
                              hipStream_t stream) {
  const float*         images = (const float*)d_in[0];         // [32][256][64][64]
  const float*         words  = (const float*)d_in[1];         // [32][32][768]
  const unsigned char* maskp  = (const unsigned char*)d_in[2]; // [32][32] bool
  const float*         Wm     = (const float*)d_in[3];         // [256][768]
  const float*         bias   = (const float*)d_in[4];         // [256]

  float* outW = (float*)d_out;                                  // [32][256][4096]
  float* outA = (float*)d_out + (size_t)32 * 256 * 4096;        // [32][32][4096]
  float* wp   = (float*)d_ws;                                   // [32][256][32]

  wordproj_kernel<<<dim3(32), dim3(256), 0, stream>>>(words, Wm, bias, wp);
  attn_kernel<<<dim3(32, 32), dim3(256), 0, stream>>>(images, wp, maskp, outW, outA);
}
